// TinyBlockReal_34849364639934
// MI455X (gfx1250) — matrix operand order repaired and checked by reading, not compiled, not run on hardware
//
#include <hip/hip_runtime.h>
#include <hip/hip_bf16.h>
#include <math.h>

// ---------------------------------------------------------------------------
// Transformer block (T=4096, C=768, FF=3072, 12 heads x 64) for MI455X gfx1250.
// ~110 GFLOP vs ~100MB traffic -> all GEMMs + attention einsums run
// v_wmma_f32_16x16x32_f16 (f16 inputs, f32 accumulate). LN/softmax/GELU f32.
// DPP16 butterfly reductions (no ds_bpermute), V staged transposed for
// contiguous PV fragment loads, async global->LDS staging (ASYNCcnt path).
// ---------------------------------------------------------------------------

typedef _Float16 half_t;
typedef __attribute__((ext_vector_type(16))) _Float16 v16h;
typedef __attribute__((ext_vector_type(8)))  float    v8f;

__device__ __forceinline__ v8f wmma_f16(v16h a, v16h b, v8f c) {
    // (neg_a, A, neg_b, B, c_mod, C, reuse_a, reuse_b)
    return __builtin_amdgcn_wmma_f32_16x16x32_f16(false, a, false, b,
                                                  (short)0, c, false, false);
}

// Load a 16-half fragment as two 16B chunks (both 16B aligned by construction).
__device__ __forceinline__ v16h ldfrag(const half_t* p0, const half_t* p1) {
    union { uint4 q[2]; v16h v; } u;
    u.q[0] = *(const uint4*)p0;
    u.q[1] = *(const uint4*)p1;
    return u.v;
}

// --------------------------- DPP16 butterfly reductions --------------------
// Reduction stays inside a 16-lane row: quad_perm xor1, xor2, half_mirror,
// mirror. Pure VALU, no LDS traffic, no DS-counter waits.
template <int CTRL>
__device__ __forceinline__ float dpp_xchg(float v) {
    int i = __builtin_bit_cast(int, v);
    int r = __builtin_amdgcn_update_dpp(i, i, CTRL, 0xF, 0xF, true);
    return __builtin_bit_cast(float, r);
}
__device__ __forceinline__ float red16_max(float v) {
    v = fmaxf(v, dpp_xchg<0xB1>(v));   // quad_perm [1,0,3,2]
    v = fmaxf(v, dpp_xchg<0x4E>(v));   // quad_perm [2,3,0,1]
    v = fmaxf(v, dpp_xchg<0x141>(v));  // row_half_mirror
    v = fmaxf(v, dpp_xchg<0x140>(v));  // row_mirror
    return v;
}
__device__ __forceinline__ float red16_sum(float v) {
    v += dpp_xchg<0xB1>(v);
    v += dpp_xchg<0x4E>(v);
    v += dpp_xchg<0x141>(v);
    v += dpp_xchg<0x140>(v);
    return v;
}

// --------------------------- async global -> LDS ---------------------------
#if defined(__has_builtin)
#if __has_builtin(__builtin_amdgcn_global_load_async_to_lds_b128)
#define HAVE_ASYNC_LDS 1
#endif
#endif
#ifndef HAVE_ASYNC_LDS
#define HAVE_ASYNC_LDS 0
#endif

// The builtin wants pointers to a 16B GCC-style int vector (per hipcc's
// diagnostic): global side in AS1 (__device__), LDS side in AS3.
typedef int async_v4i __attribute__((vector_size(16)));
typedef __attribute__((address_space(1))) async_v4i* async_gptr;
typedef __attribute__((address_space(3))) async_v4i* async_lptr;

__device__ __forceinline__ void async_copy16(const void* g, void* l) {
#if HAVE_ASYNC_LDS
    __builtin_amdgcn_global_load_async_to_lds_b128(
        (async_gptr)(g), (async_lptr)(l), 0, 0);
#else
    *(uint4*)l = *(const uint4*)g;
#endif
}
__device__ __forceinline__ void async_join() {
#if HAVE_ASYNC_LDS
    asm volatile("s_wait_asynccnt 0" ::: "memory");
#endif
}

// ---------------------------------------------------------------------------
// elementwise f32 -> f16
// ---------------------------------------------------------------------------
__global__ __launch_bounds__(256)
void cvt_f16_kernel(const float* __restrict__ s, half_t* __restrict__ d, int n) {
    int i = blockIdx.x * 256 + threadIdx.x;
    if (i < n) d[i] = (half_t)s[i];
}

// ---------------------------------------------------------------------------
// LayerNorm(row of 768) * scale -> f16.  One block per row.
// ---------------------------------------------------------------------------
__global__ __launch_bounds__(256)
void ln_to_half_kernel(const float* __restrict__ X, const float* __restrict__ g,
                       const float* __restrict__ b, half_t* __restrict__ H,
                       float scale) {
    const int C = 768;
    const int row = blockIdx.x;
    const int tid = threadIdx.x, lane = tid & 31, wave = tid >> 5;
    const float* x = X + (size_t)row * C;

    float s = 0.f, ss = 0.f;
    for (int i = tid; i < C; i += 256) { float v = x[i]; s += v; ss += v * v; }
    s  = red16_sum(s);   // per-16-lane partials (DPP, no LDS)
    ss = red16_sum(ss);

    __shared__ float rs[16], rss[16];
    __shared__ float mu_sh, rstd_sh;
    if ((lane & 15) == 0) {
        int idx = wave * 2 + (lane >> 4);
        rs[idx] = s; rss[idx] = ss;
    }
    __syncthreads();
    if (tid == 0) {
        float S = 0.f, SS = 0.f;
#pragma unroll
        for (int i = 0; i < 16; ++i) { S += rs[i]; SS += rss[i]; }
        float mu  = S / C;
        float var = SS / C - mu * mu;
        mu_sh = mu;
        rstd_sh = rsqrtf(var + 1e-5f);
    }
    __syncthreads();
    const float mu = mu_sh, rstd = rstd_sh;
    for (int i = tid; i < C; i += 256) {
        float v = (x[i] - mu) * rstd * g[i] + b[i];
        H[(size_t)row * C + i] = (half_t)(v * scale);
    }
}

// ---------------------------------------------------------------------------
// WMMA GEMM:  C[M,N] = A[M,K](f16) * B[K,N](f16) + bias, with epilogue:
//   EPI 0: store f16
//   EPI 1: exact-erf GELU, store f16
//   EPI 2: f32 out = residual + acc + bias
// Block tile 64x128, 8 waves (2x4), wave tile 32x32 (2x2 WMMA frags), K-step 32.
// A tile staged with async-to-LDS; B tile staged transposed (scatter stores),
// so all fragment loads are contiguous ds_load_b128s.
// ---------------------------------------------------------------------------
template <int EPI>
__global__ __launch_bounds__(256)
void gemm_wmma_kernel(const half_t* __restrict__ A, const half_t* __restrict__ B,
                      const float* __restrict__ bias, const float* __restrict__ R,
                      half_t* __restrict__ O16, float* __restrict__ O32,
                      int M, int N, int K) {
    __shared__ half_t sA[64][32];    // [m][k]
    __shared__ half_t sBt[128][32];  // [n][k] (transposed)

    const int tid  = threadIdx.x;
    const int lane = tid & 31;
    const int wave = tid >> 5;
    const int wr   = wave >> 2;   // 0..1  (wave row: 32 rows each)
    const int wc   = wave & 3;    // 0..3  (wave col: 32 cols each)
    const int m0   = blockIdx.y * 64;
    const int n0   = blockIdx.x * 128;
    const int r16  = lane & 15;
    const int hi   = lane >> 4;

    v8f acc[2][2] = {};

    for (int k0 = 0; k0 < K; k0 += 32) {
        // A tile: 64x32 halves -> one async 16B per thread, direct to LDS
        {
            int r = tid >> 2;
            int cg = (tid & 3) * 8;
            async_copy16(&A[(size_t)(m0 + r) * K + k0 + cg], &sA[r][cg]);
        }
        // B tile transposed (overlaps with async A): coalesced uint4 read
        // along N, scatter 8 u16 stores into LDS.
#pragma unroll
        for (int s = 0; s < 2; ++s) {
            int task = tid * 2 + s;
            int kk = task & 31;
            int ng = (task >> 5) * 8;
            uint4 d = *(const uint4*)&B[(size_t)(k0 + kk) * N + n0 + ng];
            const half_t* hv = (const half_t*)&d;
#pragma unroll
            for (int i = 0; i < 8; ++i) sBt[ng + i][kk] = hv[i];
        }
        async_join();
        __syncthreads();

        // A frags: lane = row (0-15), halves k = hi*8 + {0..7, 16..23}
        v16h af[2], bf[2];
#pragma unroll
        for (int m = 0; m < 2; ++m) {
            const half_t* p = &sA[wr * 32 + m * 16 + r16][hi * 8];
            af[m] = ldfrag(p, p + 16);
        }
        // B frags (from transposed tile): lane = col, k = hi*8 + {0..7, 16..23}
#pragma unroll
        for (int n = 0; n < 2; ++n) {
            const half_t* p = &sBt[wc * 32 + n * 16 + r16][hi * 8];
            bf[n] = ldfrag(p, p + 16);
        }
#pragma unroll
        for (int m = 0; m < 2; ++m)
#pragma unroll
            for (int n = 0; n < 2; ++n)
                acc[m][n] = wmma_f16(af[m], bf[n], acc[m][n]);
        __syncthreads();
    }

    // epilogue. D layout: VGPR r, lanes0-15 -> M=r, lanes16-31 -> M=8+r
#pragma unroll
    for (int m = 0; m < 2; ++m) {
#pragma unroll
        for (int n = 0; n < 2; ++n) {
            const int col = n0 + wc * 32 + n * 16 + r16;
            const int rbase = m0 + wr * 32 + m * 16 + hi * 8;
            const float bv = bias[col];
#pragma unroll
            for (int r = 0; r < 8; ++r) {
                const size_t idx = (size_t)(rbase + r) * N + col;
                float v = acc[m][n][r] + bv;
                if (EPI == 0) {
                    O16[idx] = (half_t)v;
                } else if (EPI == 1) {
                    float ge = 0.5f * v * (1.0f + erff(v * 0.70710678118654752f));
                    O16[idx] = (half_t)ge;
                } else {
                    O32[idx] = R[idx] + v;
                }
            }
        }
    }
}

// ---------------------------------------------------------------------------
// Flash attention (no mask). QKV packed f16 [T, 2304] (q|k|v per head of 64).
// Block = 8 waves handles (head, 128 query rows); each wave owns a 16-row
// query stripe. Streaming softmax with DPP reductions; scores & PV via WMMA.
// K staged natural (async), V staged transposed for contiguous B-frag loads.
// ---------------------------------------------------------------------------
__global__ __launch_bounds__(256)
void attn_wmma_kernel(const half_t* __restrict__ QKV, half_t* __restrict__ O,
                      int T) {
    const int C3 = 2304;
    __shared__ half_t sK[32][64];     // [key][hd]
    __shared__ half_t sVt[64][32];    // [hd][key]  (transposed)
    __shared__ half_t sP[8][16][32];  // per-wave prob patch (D -> A layout)

    const int head = blockIdx.y;
    const int q0   = blockIdx.x * 128;
    const int tid  = threadIdx.x, lane = tid & 31, wave = tid >> 5;
    const int r16  = lane & 15, hi = lane >> 4;
    const int qoff = head * 64, koff = 768 + head * 64, voff = 1536 + head * 64;

    // Q fragments (16 rows x 64 hd = two 16x32 A-frags), straight from global
    const int qrow = q0 + wave * 16 + r16;
    v16h qa[2];
#pragma unroll
    for (int c = 0; c < 2; ++c) {
        const half_t* p = &QKV[(size_t)qrow * C3 + qoff + c * 32 + hi * 8];
        qa[c] = ldfrag(p, p + 16);
    }

    v8f o[4] = {};
    float mrun[8], lrun[8];
#pragma unroll
    for (int r = 0; r < 8; ++r) { mrun[r] = -1e30f; lrun[r] = 0.f; }

    for (int kt = 0; kt < T; kt += 32) {
        // stage K (async, natural layout) and V (transposed scatter)
        {
            int r = tid >> 3, cg = (tid & 7) * 8;
            async_copy16(&QKV[(size_t)(kt + r) * C3 + koff + cg], &sK[r][cg]);
            uint4 d = *(const uint4*)&QKV[(size_t)(kt + r) * C3 + voff + cg];
            const half_t* hv = (const half_t*)&d;
#pragma unroll
            for (int i = 0; i < 8; ++i) sVt[cg + i][r] = hv[i];
        }
        async_join();
        __syncthreads();

        // scores S[16q x 32k] = Q * K^T, two 16-key D frags, 2 WMMAs each
        v8f sc[2];
#pragma unroll
        for (int j = 0; j < 2; ++j) {
            v8f a = {};
#pragma unroll
            for (int c = 0; c < 2; ++c) {
                const half_t* p = &sK[j * 16 + r16][c * 32 + hi * 8];
                v16h bf = ldfrag(p, p + 16);
                a = wmma_f16(qa[c], bf, a);
            }
            sc[j] = a;
        }
#pragma unroll
        for (int j = 0; j < 2; ++j)
#pragma unroll
            for (int r = 0; r < 8; ++r) sc[j][r] *= 0.125f;  // 1/sqrt(64)

        // streaming softmax; row N-values live across a 16-lane half (DPP)
        float mnew[8], alpha[8], psum[8];
#pragma unroll
        for (int r = 0; r < 8; ++r) {
            float mx = red16_max(fmaxf(sc[0][r], sc[1][r]));
            mnew[r]  = fmaxf(mrun[r], mx);
            alpha[r] = __expf(mrun[r] - mnew[r]);
            psum[r]  = 0.f;
        }
#pragma unroll
        for (int j = 0; j < 2; ++j)
#pragma unroll
            for (int r = 0; r < 8; ++r) {
                float p = __expf(sc[j][r] - mnew[r]);
                sc[j][r] = p;
                psum[r] += p;
            }
#pragma unroll
        for (int r = 0; r < 8; ++r) {
            psum[r] = red16_sum(psum[r]);
            lrun[r] = lrun[r] * alpha[r] + psum[r];
            mrun[r] = mnew[r];
        }
#pragma unroll
        for (int n = 0; n < 4; ++n)
#pragma unroll
            for (int r = 0; r < 8; ++r) o[n][r] *= alpha[r];

        // bounce probs through LDS: D-layout -> A-layout for the PV WMMA
#pragma unroll
        for (int j = 0; j < 2; ++j)
#pragma unroll
            for (int r = 0; r < 8; ++r)
                sP[wave][r + 8 * hi][j * 16 + r16] = (half_t)sc[j][r];
        asm volatile("s_wait_dscnt 0" ::: "memory");
        const half_t* pp = &sP[wave][r16][hi * 8];
        v16h pa = ldfrag(pp, pp + 16);

        // O += P[16x32] * V[32x64]; B frags contiguous from transposed V
#pragma unroll
        for (int n = 0; n < 4; ++n) {
            const half_t* p = &sVt[n * 16 + r16][hi * 8];
            v16h bv = ldfrag(p, p + 16);
            o[n] = wmma_f16(pa, bv, o[n]);
        }
        __syncthreads();
    }

    // finalize: divide by running sum, store f16 [T, 768]
    float inv[8];
#pragma unroll
    for (int r = 0; r < 8; ++r) inv[r] = 1.0f / lrun[r];
#pragma unroll
    for (int n = 0; n < 4; ++n)
#pragma unroll
        for (int r = 0; r < 8; ++r) {
            const int row = q0 + wave * 16 + r + 8 * hi;
            const int col = head * 64 + n * 16 + r16;
            O[(size_t)row * 768 + col] = (half_t)(o[n][r] * inv[r]);
        }
}

// ---------------------------------------------------------------------------
// host-side launcher
// ---------------------------------------------------------------------------
extern "C" void kernel_launch(void* const* d_in, const int* in_sizes, int n_in,
                              void* d_out, int out_size, void* d_ws, size_t ws_size,
                              hipStream_t stream) {
    (void)in_sizes; (void)n_in; (void)out_size; (void)ws_size;
    const int T = 4096, C = 768, C3 = 2304, FF = 3072;

    const float* x     = (const float*)d_in[0];
    const float* ln1_g = (const float*)d_in[1];
    const float* ln1_b = (const float*)d_in[2];
    const float* ln2_g = (const float*)d_in[3];
    const float* ln2_b = (const float*)d_in[4];
    const float* qkv_w = (const float*)d_in[5];
    const float* qkv_b = (const float*)d_in[6];
    const float* out_w = (const float*)d_in[7];
    const float* out_b = (const float*)d_in[8];
    const float* fc1_w = (const float*)d_in[9];
    const float* fc1_b = (const float*)d_in[10];
    const float* fc2_w = (const float*)d_in[11];
    const float* fc2_b = (const float*)d_in[12];
    float* out = (float*)d_out;

    // workspace carve (256B aligned)
    char* wsp = (char*)d_ws;
    auto carve = [&](size_t bytes) {
        void* p = (void*)wsp;
        wsp += (bytes + 255) & ~(size_t)255;
        return p;
    };
    half_t* h16    = (half_t*)carve((size_t)T * C  * 2);
    half_t* qkv16  = (half_t*)carve((size_t)T * C3 * 2);
    half_t* attn16 = (half_t*)carve((size_t)T * C  * 2);
    float*  x1     = (float*) carve((size_t)T * C  * 4);
    half_t* h2     = (half_t*)carve((size_t)T * C  * 2);
    half_t* ff16   = (half_t*)carve((size_t)T * FF * 2);
    half_t* wqkv   = (half_t*)carve((size_t)C  * C3 * 2);
    half_t* wout   = (half_t*)carve((size_t)C  * C  * 2);
    half_t* wfc1   = (half_t*)carve((size_t)C  * FF * 2);
    half_t* wfc2   = (half_t*)carve((size_t)FF * C  * 2);

    auto cvt = [&](const float* s, half_t* d, int n) {
        cvt_f16_kernel<<<(n + 255) / 256, 256, 0, stream>>>(s, d, n);
    };
    cvt(qkv_w, wqkv, C * C3);
    cvt(out_w, wout, C * C);
    cvt(fc1_w, wfc1, C * FF);
    cvt(fc2_w, wfc2, FF * C);

    // --- attention sub-block ---
    ln_to_half_kernel<<<T, 256, 0, stream>>>(x, ln1_g, ln1_b, h16, 0.5f);

    gemm_wmma_kernel<0><<<dim3(C3 / 128, T / 64), 256, 0, stream>>>(
        h16, wqkv, qkv_b, nullptr, qkv16, nullptr, T, C3, C);

    attn_wmma_kernel<<<dim3(T / 128, 12), 256, 0, stream>>>(qkv16, attn16, T);

    gemm_wmma_kernel<2><<<dim3(C / 128, T / 64), 256, 0, stream>>>(
        attn16, wout, out_b, x, nullptr, x1, T, C, C);

    // --- MLP sub-block ---
    ln_to_half_kernel<<<T, 256, 0, stream>>>(x1, ln2_g, ln2_b, h2, 0.5f);

    gemm_wmma_kernel<1><<<dim3(FF / 128, T / 64), 256, 0, stream>>>(
        h2, wfc1, fc1_b, nullptr, ff16, nullptr, T, FF, C);

    gemm_wmma_kernel<2><<<dim3(C / 128, T / 64), 256, 0, stream>>>(
        ff16, wfc2, fc2_b, x1, nullptr, out, T, C, FF);
}
